// GraphSAGE_75926431859108
// MI455X (gfx1250) — matrix operand order repaired and checked by reading, not compiled, not run on hardware
//
#include <hip/hip_runtime.h>
#include <hip/hip_bf16.h>

#define N_NODES  50000
#define N_EDGES  800000
#define N_GRAPHS 128
#define DIM      128
#define CLS      10

typedef __attribute__((ext_vector_type(16))) __bf16       v16bf;
typedef __attribute__((ext_vector_type(8)))  float        v8f;
typedef __attribute__((ext_vector_type(8)))  unsigned int v8u;

static __device__ __forceinline__ unsigned int f2bf_pack(float lo, float hi) {
    unsigned int ul = __float_as_uint(lo);
    unsigned int uh = __float_as_uint(hi);
    ul = (ul + 0x7FFFu + ((ul >> 16) & 1u)) >> 16;   // round-to-nearest-even
    uh = (uh + 0x7FFFu + ((uh >> 16) & 1u)) >> 16;
    return (ul & 0xFFFFu) | (uh << 16);
}

// ---------------------------------------------------------------- utilities
__global__ void fill0_kernel(float* __restrict__ p, int n) {
    int i = blockIdx.x * blockDim.x + threadIdx.x;
    if (i < n) p[i] = 0.0f;
}

__global__ void deg_kernel(const int* __restrict__ dst, float* __restrict__ deg) {
    int e = blockIdx.x * blockDim.x + threadIdx.x;
    if (e < N_EDGES) atomicAdd(&deg[dst[e]], 1.0f);
}

__global__ void invdeg_kernel(float* __restrict__ deg) {
    int i = blockIdx.x * blockDim.x + threadIdx.x;
    if (i < N_NODES) deg[i] = 1.0f / fmaxf(deg[i], 1.0f);
}

// one wave (32 threads) per edge, each thread handles 4 consecutive dims
__global__ void scatter_kernel(const float* __restrict__ h,
                               const int* __restrict__ src,
                               const int* __restrict__ dst,
                               float* __restrict__ msg) {
    long long t = (long long)blockIdx.x * blockDim.x + threadIdx.x;
    int e = (int)(t >> 5);
    if (e >= N_EDGES) return;
    int d = ((int)t & 31) << 2;
    int s = src[e], q = dst[e];
    const float4 v = *(const float4*)(h + (size_t)s * DIM + d);
    float* mp = msg + (size_t)q * DIM + d;
    atomicAdd(mp + 0, v.x);
    atomicAdd(mp + 1, v.y);
    atomicAdd(mp + 2, v.z);
    atomicAdd(mp + 3, v.w);
}

// ------------------------------------------------------------- fused GEMM
// hout[row][:] = relu( hin[row]@Ws + (msg[row]*invdeg[row])@Wn + bias )
// block = 256 threads = 8 waves; wave w owns rows [blk*128 + 16w, +16)
__global__ __launch_bounds__(256) void sage_gemm_kernel(
        const float* __restrict__ hin,  const float* __restrict__ msg,
        const float* __restrict__ invdeg,
        const float* __restrict__ Ws,   const float* __restrict__ Wn,
        const float* __restrict__ bias, float* __restrict__ hout) {
    // W transposed + packed to bf16 pairs: WtLds[n*64 + k/2] = (W[k][n], W[k+1][n])
    __shared__ unsigned int WtLds[DIM * 64];   // 32 KB

    const int tid  = threadIdx.x;
    const int wave = tid >> 5;
    const int lane = tid & 31;
    const int hi16 = lane >> 4;          // 0 or 1 (lane half)
    const int lmod = lane & 15;
    const int rowbase = blockIdx.x * 128 + wave * 16;
    const int arow    = rowbase + lmod;                      // A-matrix row (M = lane&15)
    const int arow_c  = arow < N_NODES ? arow : (N_NODES - 1);
    const float idg   = invdeg[arow_c];
    const int klane   = hi16 * 8;        // A elems 0..7 -> K = kb..kb+7

    v8f acc[8];
#pragma unroll
    for (int nt = 0; nt < 8; ++nt) acc[nt] = {};

    auto stage = [&](const float* __restrict__ W) {
        for (int idx = tid; idx < DIM * 64; idx += 256) {
            int n = idx >> 6;
            int k = (idx & 63) << 1;
            WtLds[idx] = f2bf_pack(W[k * DIM + n], W[(k + 1) * DIM + n]);
        }
    };

    auto do_phase = [&](const float* __restrict__ A, bool scale) {
#pragma unroll
        for (int s = 0; s < 4; ++s) {
            const int kb = s * 32 + klane;
            const float* ap = A + (size_t)arow_c * DIM + kb;
            float4 f0 = *(const float4*)(ap);
            float4 f1 = *(const float4*)(ap + 4);
            float4 f2 = *(const float4*)(ap + 16);
            float4 f3 = *(const float4*)(ap + 20);
            if (scale) {
                f0.x *= idg; f0.y *= idg; f0.z *= idg; f0.w *= idg;
                f1.x *= idg; f1.y *= idg; f1.z *= idg; f1.w *= idg;
                f2.x *= idg; f2.y *= idg; f2.z *= idg; f2.w *= idg;
                f3.x *= idg; f3.y *= idg; f3.z *= idg; f3.w *= idg;
            }
            v8u au;
            au[0] = f2bf_pack(f0.x, f0.y);
            au[1] = f2bf_pack(f0.z, f0.w);
            au[2] = f2bf_pack(f1.x, f1.y);
            au[3] = f2bf_pack(f1.z, f1.w);
            au[4] = f2bf_pack(f2.x, f2.y);
            au[5] = f2bf_pack(f2.z, f2.w);
            au[6] = f2bf_pack(f3.x, f3.y);
            au[7] = f2bf_pack(f3.z, f3.w);
            v16bf afrag = __builtin_bit_cast(v16bf, au);

            const int kpair = (s * 32 + hi16 * 8) >> 1;    // B: K base / 2
#pragma unroll
            for (int nt = 0; nt < 8; ++nt) {
                const unsigned int* bp = &WtLds[(nt * 16 + lmod) * 64 + kpair];
                v8u bu;
#pragma unroll
                for (int j = 0; j < 8; ++j) bu[j] = bp[j < 4 ? j : j + 4];
                v16bf bfrag = __builtin_bit_cast(v16bf, bu);
                acc[nt] = __builtin_amdgcn_wmma_f32_16x16x32_bf16(
                    false, afrag, false, bfrag, (short)0, acc[nt], false, false);
            }
        }
    };

    stage(Ws);
    __syncthreads();
    do_phase(hin, false);
    __syncthreads();
    stage(Wn);
    __syncthreads();
    do_phase(msg, true);

    // C/D layout: elem r of lane l -> row = rowbase + r + 8*(l>>4), col = l&15
#pragma unroll
    for (int nt = 0; nt < 8; ++nt) {
        const int col = nt * 16 + lmod;
        const float bb = bias[col];
#pragma unroll
        for (int r = 0; r < 8; ++r) {
            int row = rowbase + r + hi16 * 8;
            if (row < N_NODES) {
                float v = acc[nt][r] + bb;
                hout[(size_t)row * DIM + col] = v > 0.0f ? v : 0.0f;
            }
        }
    }
}

// ---------------------------------------------------------------- pooling
__global__ void count_kernel(const int* __restrict__ gid, float* __restrict__ counts) {
    int i = blockIdx.x * blockDim.x + threadIdx.x;
    if (i < N_NODES) atomicAdd(&counts[gid[i]], 1.0f);
}

__global__ void pool_kernel(const float* __restrict__ h,
                            const int* __restrict__ gid,
                            float* __restrict__ pooled) {
    long long t = (long long)blockIdx.x * blockDim.x + threadIdx.x;
    int n = (int)(t >> 5);
    if (n >= N_NODES) return;
    int d = ((int)t & 31) << 2;
    int g = gid[n];
    const float4 v = *(const float4*)(h + (size_t)n * DIM + d);
    float* p = pooled + (size_t)g * DIM + d;
    atomicAdd(p + 0, v.x);
    atomicAdd(p + 1, v.y);
    atomicAdd(p + 2, v.z);
    atomicAdd(p + 3, v.w);
}

__global__ void final_kernel(const float* __restrict__ pooled,
                             const float* __restrict__ counts,
                             const float* __restrict__ Wf,
                             const float* __restrict__ bf,
                             float* __restrict__ out) {
    __shared__ float s[DIM];
    int g = blockIdx.x;
    int d = threadIdx.x;
    float inv = 1.0f / fmaxf(counts[g], 1.0f);
    s[d] = pooled[(size_t)g * DIM + d] * inv;
    __syncthreads();
    if (d < CLS) {
        float acc = bf[d];
#pragma unroll 4
        for (int k = 0; k < DIM; ++k) acc += s[k] * Wf[k * CLS + d];
        out[g * CLS + d] = acc;
    }
}

// ---------------------------------------------------------------- launcher
static inline int cdiv(long long a, int b) { return (int)((a + b - 1) / b); }

extern "C" void kernel_launch(void* const* d_in, const int* in_sizes, int n_in,
                              void* d_out, int out_size, void* d_ws, size_t ws_size,
                              hipStream_t stream) {
    (void)in_sizes; (void)n_in; (void)out_size; (void)ws_size;

    const float* features = (const float*)d_in[0];
    const int*   esrc     = (const int*)d_in[1];
    const int*   edst     = (const int*)d_in[2];
    const int*   gid      = (const int*)d_in[3];
    const float* Ws[3]    = {(const float*)d_in[4], (const float*)d_in[7], (const float*)d_in[10]};
    const float* Wn[3]    = {(const float*)d_in[5], (const float*)d_in[8], (const float*)d_in[11]};
    const float* bs[3]    = {(const float*)d_in[6], (const float*)d_in[9], (const float*)d_in[12]};
    const float* Wf       = (const float*)d_in[13];
    const float* bfin     = (const float*)d_in[14];
    float*       out      = (float*)d_out;

    // workspace carve-up (floats)
    float* ws      = (float*)d_ws;
    float* invdeg  = ws;                                 // 50000
    float* msg     = ws + 65536;                         // 6,400,000
    float* hA      = msg + (size_t)N_NODES * DIM;        // 6,400,000
    float* hB      = hA  + (size_t)N_NODES * DIM;        // 6,400,000
    float* pooled  = hB  + (size_t)N_NODES * DIM;        // 16384
    float* counts  = pooled + (size_t)N_GRAPHS * DIM;    // 128

    // in-degree -> 1/max(deg,1)
    fill0_kernel<<<cdiv(N_NODES, 256), 256, 0, stream>>>(invdeg, N_NODES);
    deg_kernel<<<cdiv(N_EDGES, 256), 256, 0, stream>>>(edst, invdeg);
    invdeg_kernel<<<cdiv(N_NODES, 256), 256, 0, stream>>>(invdeg);

    const int scatter_blocks = cdiv((long long)N_EDGES * 32, 256);
    const int gemm_blocks    = cdiv(N_NODES, 128);

    const float* hin = features;
    float* houts[3] = {hA, hB, hA};
    for (int l = 0; l < 3; ++l) {
        fill0_kernel<<<cdiv((long long)N_NODES * DIM, 256), 256, 0, stream>>>(msg, N_NODES * DIM);
        scatter_kernel<<<scatter_blocks, 256, 0, stream>>>(hin, esrc, edst, msg);
        sage_gemm_kernel<<<gemm_blocks, 256, 0, stream>>>(
            hin, msg, invdeg, Ws[l], Wn[l], bs[l], houts[l]);
        hin = houts[l];
    }

    // mean pooling + classifier head
    fill0_kernel<<<cdiv(N_GRAPHS * DIM, 256), 256, 0, stream>>>(pooled, N_GRAPHS * DIM);
    fill0_kernel<<<1, 256, 0, stream>>>(counts, N_GRAPHS);
    count_kernel<<<cdiv(N_NODES, 256), 256, 0, stream>>>(gid, counts);
    pool_kernel<<<cdiv((long long)N_NODES * 32, 256), 256, 0, stream>>>(hin, gid, pooled);
    final_kernel<<<N_GRAPHS, DIM, 0, stream>>>(pooled, counts, Wf, bfin, out);
}
